// EuclideanCodebook_15255723835629
// MI455X (gfx1250) — compile-verified, matrix operand order checked by reading, not run on hardware
//
#include <hip/hip_runtime.h>
#include <hip/hip_bf16.h>
#include <stdint.h>

// ---------------------------------------------------------------------------
// EuclideanCodebook for MI455X (gfx1250, wave32, WMMA)
//   encode: argmax_n ( 2 * x.e_n - ||e_n||^2 )   (||x||^2 dropped: row-const)
//   decode: quantize = embed[idx]
// GEMM via v_wmma_f32_16x16x32_bf16 with hi/lo bf16 split of fp32 inputs
// (3 WMMAs per tile: hi*hi + hi*lo + lo*hi) for ~fp32 ranking accuracy.
// Hot loop: double-buffered LDS tiles filled with GLOBAL_LOAD_ASYNC_TO_LDS_B128
// (ASYNCcnt), software-pipelined so the DMA for stage s+1 overlaps the WMMAs
// of stage s.  Fallback kernel converts inline if workspace is too small.
// ---------------------------------------------------------------------------

typedef __attribute__((ext_vector_type(16))) __bf16 v16bf;
typedef __attribute__((ext_vector_type(8)))  float  v8f;

#define DDIM    512
#define BM      128
#define BN      128
#define BK      32
#define BKP     40            // padded LDS row stride (80B: conflict-free)
#define KSTAGES (DDIM / BK)   // 16 k-steps per N-tile

__device__ __forceinline__ uint16_t f32_to_bf16_bits(float f) {
  union { float f; uint32_t u; } c; c.f = f;
  uint32_t u = c.u;
  uint32_t r = u + 0x7FFFu + ((u >> 16) & 1u);   // round-to-nearest-even
  return (uint16_t)(r >> 16);
}
__device__ __forceinline__ float bf16_bits_to_f32(uint16_t h) {
  union { uint32_t u; float f; } c; c.u = ((uint32_t)h) << 16;
  return c.f;
}

// ---------------------------------------------------------------------------
// Kernel 0: fp32 -> packed bf16 hi/lo planes (8 floats per thread).
// ---------------------------------------------------------------------------
__global__ void convert_hilo_kernel(const float* __restrict__ src,
                                    uint16_t* __restrict__ hi,
                                    uint16_t* __restrict__ lo,
                                    int n8) {
  int g = blockIdx.x * blockDim.x + threadIdx.x;
  if (g >= n8) return;
  const float4* s = (const float4*)src + (size_t)g * 2;
  float4 a = s[0], b = s[1];
  float f[8] = {a.x, a.y, a.z, a.w, b.x, b.y, b.z, b.w};
  uint32_t ph[4], pl[4];
  #pragma unroll
  for (int i = 0; i < 4; ++i) {
    uint16_t h0 = f32_to_bf16_bits(f[2 * i]);
    uint16_t h1 = f32_to_bf16_bits(f[2 * i + 1]);
    uint16_t l0 = f32_to_bf16_bits(f[2 * i]     - bf16_bits_to_f32(h0));
    uint16_t l1 = f32_to_bf16_bits(f[2 * i + 1] - bf16_bits_to_f32(h1));
    ph[i] = (uint32_t)h0 | ((uint32_t)h1 << 16);
    pl[i] = (uint32_t)l0 | ((uint32_t)l1 << 16);
  }
  ((uint4*)hi)[g] = make_uint4(ph[0], ph[1], ph[2], ph[3]);
  ((uint4*)lo)[g] = make_uint4(pl[0], pl[1], pl[2], pl[3]);
}

// ---------------------------------------------------------------------------
// Kernel 1: e_sq[c] = sum_j embed[c][j]^2     (grid = K, block = 128)
// ---------------------------------------------------------------------------
__global__ void esq_kernel(const float* __restrict__ embed,
                           float* __restrict__ e_sq) {
  const int c = blockIdx.x;
  const int t = threadIdx.x;            // 0..127
  float s = 0.0f;
  #pragma unroll
  for (int j = 0; j < DDIM; j += 128) {
    float v = embed[(size_t)c * DDIM + j + t];
    s += v * v;
  }
  #pragma unroll
  for (int off = 16; off >= 1; off >>= 1)
    s += __shfl_xor(s, off, 32);
  __shared__ float red[4];
  if ((t & 31) == 0) red[t >> 5] = s;
  __syncthreads();
  if (t == 0) e_sq[c] = red[0] + red[1] + red[2] + red[3];
}

// ---------------------------------------------------------------------------
// Async tile stage: 128x32 bf16 plane pair -> LDS via
// GLOBAL_LOAD_ASYNC_TO_LDS_B128 (ASYNCcnt).  INST_OFFSET is added to both the
// global and LDS addresses (ISA §10 async pseudocode), so the two 16B chunks
// per plane share base registers via offset:16.  4 instructions issued.
// 256 threads: thread t handles row t>>1, 16-element half (t&1).
// ---------------------------------------------------------------------------
__device__ __forceinline__ void load_tile_async(const uint16_t* __restrict__ hi_g,
                                                const uint16_t* __restrict__ lo_g,
                                                int rowBase, int kk,
                                                uint16_t* __restrict__ hiS,
                                                uint16_t* __restrict__ loS) {
  const int t    = threadIdx.x;
  const int row  = t >> 1;
  const int half = t & 1;
  const size_t off  = (size_t)(rowBase + row) * DDIM + kk + half * 16;
  const int    soff = row * BKP + half * 16;
  uint64_t gh = (uint64_t)(uintptr_t)(hi_g + off);
  uint64_t gl = (uint64_t)(uintptr_t)(lo_g + off);
  // Low 32 bits of a generic LDS pointer = LDS byte offset (flat aperture).
  uint32_t dh = (uint32_t)(uintptr_t)(hiS + soff);
  uint32_t dl = (uint32_t)(uintptr_t)(loS + soff);
  asm volatile(
      "global_load_async_to_lds_b128 %0, %1, off\n\t"
      "global_load_async_to_lds_b128 %0, %1, off offset:16\n\t"
      "global_load_async_to_lds_b128 %2, %3, off\n\t"
      "global_load_async_to_lds_b128 %2, %3, off offset:16"
      :: "v"(dh), "v"(gh), "v"(dl), "v"(gl)
      : "memory");
}

// Fallback: convert fp32 -> bf16 hi/lo inline (synchronous loads).
__device__ __forceinline__ void load_tile_hilo(const float* __restrict__ g,
                                               int rowBase, int kk,
                                               uint16_t* __restrict__ hi,
                                               uint16_t* __restrict__ lo) {
  const int t    = threadIdx.x;
  const int row  = t >> 1;
  const int half = t & 1;
  const float4* src =
      (const float4*)(g + (size_t)(rowBase + row) * DDIM + kk + half * 16);
  float4 v0 = src[0], v1 = src[1], v2 = src[2], v3 = src[3];
  float f[16] = {v0.x, v0.y, v0.z, v0.w, v1.x, v1.y, v1.z, v1.w,
                 v2.x, v2.y, v2.z, v2.w, v3.x, v3.y, v3.z, v3.w};
  uint32_t* dhi = (uint32_t*)(hi + row * BKP + half * 16);
  uint32_t* dlo = (uint32_t*)(lo + row * BKP + half * 16);
  #pragma unroll
  for (int i = 0; i < 8; ++i) {
    uint16_t h0 = f32_to_bf16_bits(f[2 * i]);
    uint16_t h1 = f32_to_bf16_bits(f[2 * i + 1]);
    uint16_t l0 = f32_to_bf16_bits(f[2 * i]     - bf16_bits_to_f32(h0));
    uint16_t l1 = f32_to_bf16_bits(f[2 * i + 1] - bf16_bits_to_f32(h1));
    dhi[i] = (uint32_t)h0 | ((uint32_t)h1 << 16);
    dlo[i] = (uint32_t)l0 | ((uint32_t)l1 << 16);
  }
}

// Fragment load in contiguous-K order (same for A and B, not the ISA table order):
// lane&15 -> matrix row, lane>>4 -> K half (K 0..15 | 16..31), 32B contiguous.
__device__ __forceinline__ v16bf load_frag(const uint16_t* __restrict__ base,
                                           int row16) {
  const int lane = threadIdx.x & 31;
  const uint16_t* p = base + (row16 + (lane & 15)) * BKP + (lane >> 4) * 16;
  return *reinterpret_cast<const v16bf*>(p);
}

// ---------------------------------------------------------------------------
// Kernel 2a (preferred): software-pipelined, double-buffered GEMM + argmax.
// grid = Nrows/128, block = 256 (8 waves); wave w owns 16 rows, sweeps all K.
// Stage s = (nTile, kStep); async copy for s+1 overlaps compute of s.
// ---------------------------------------------------------------------------
__global__ void
__launch_bounds__(256)
argmax_pre_kernel(const uint16_t* __restrict__ xhi,
                  const uint16_t* __restrict__ xlo,
                  const uint16_t* __restrict__ ehi,
                  const uint16_t* __restrict__ elo,
                  const float* __restrict__ e_sq,
                  int* __restrict__ idx_out,
                  float* __restrict__ ind_f,
                  int K) {
  __shared__ uint16_t sAhi[2][BM * BKP];
  __shared__ uint16_t sAlo[2][BM * BKP];
  __shared__ uint16_t sBhi[2][BN * BKP];
  __shared__ uint16_t sBlo[2][BN * BKP];

  const int w        = threadIdx.x >> 5;   // wave 0..7
  const int lane     = threadIdx.x & 31;
  const int rowFrag  = w * 16;
  const int blockRow = blockIdx.x * BM;
  const int S        = (K / BN) * KSTAGES; // total stages

  float bestVal[8];
  int   bestIdx[8];
  #pragma unroll
  for (int r = 0; r < 8; ++r) { bestVal[r] = -3.4e38f; bestIdx[r] = 0; }

  // Prologue: stage 0 into buffer 0.
  load_tile_async(xhi, xlo, blockRow, 0, sAhi[0], sAlo[0]);
  load_tile_async(ehi, elo, 0,        0, sBhi[0], sBlo[0]);

  v8f acc[8];
  for (int s = 0; s < S; ++s) {
    const int par = s & 1;
    const int ks  = s & (KSTAGES - 1);
    const int nB  = (s / KSTAGES) * BN;

    if (ks == 0) {
      #pragma unroll
      for (int t = 0; t < 8; ++t) acc[t] = (v8f){};
    }

    // Issue stage s+1 into the other buffer (safe: its previous contents were
    // consumed at stage s-1, and the trailing barrier of s-1 ordered that).
    if (s + 1 < S) {
      const int s1  = s + 1;
      const int kk1 = (s1 & (KSTAGES - 1)) * BK;
      const int nB1 = (s1 / KSTAGES) * BN;
      load_tile_async(xhi, xlo, blockRow, kk1, sAhi[par ^ 1], sAlo[par ^ 1]);
      load_tile_async(ehi, elo, nB1,      kk1, sBhi[par ^ 1], sBlo[par ^ 1]);
      // 8 async ops outstanding for s+1; stage-s ops (in-order) are done.
      asm volatile("s_wait_asynccnt 0x8" ::: "memory");
    } else {
      asm volatile("s_wait_asynccnt 0x0" ::: "memory");
    }
    __syncthreads();   // stage-s tiles visible from all waves

    v16bf ah = load_frag(sAhi[par], rowFrag);
    v16bf al = load_frag(sAlo[par], rowFrag);
    #pragma unroll
    for (int t = 0; t < 8; ++t) {
      v16bf bh = load_frag(sBhi[par], t * 16);
      v16bf bl = load_frag(sBlo[par], t * 16);
      // dot ~= lo*hi + hi*lo + hi*hi (chained through C operand)
      acc[t] = __builtin_amdgcn_wmma_f32_16x16x32_bf16(
          false, al, false, bh, (short)0, acc[t], false, false);
      acc[t] = __builtin_amdgcn_wmma_f32_16x16x32_bf16(
          false, ah, false, bl, (short)0, acc[t], false, false);
      acc[t] = __builtin_amdgcn_wmma_f32_16x16x32_bf16(
          false, ah, false, bh, (short)0, acc[t], false, false);
    }

    if (ks == KSTAGES - 1) {
      // Epilogue: score = 2*dot - ||e||^2 ; per-lane running argmax.
      #pragma unroll
      for (int t = 0; t < 8; ++t) {
        const int n    = nB + t * 16 + (lane & 15);
        const float es = e_sq[n];
        #pragma unroll
        for (int r = 0; r < 8; ++r) {
          float sc = 2.0f * acc[t][r] - es;
          if (sc > bestVal[r]) { bestVal[r] = sc; bestIdx[r] = n; }
        }
      }
    }
    __syncthreads();   // all waves done reading buf[par] before reuse
  }

  // Cross-lane reduce inside each 16-lane half (halves hold different M rows).
  #pragma unroll
  for (int r = 0; r < 8; ++r) {
    float bv = bestVal[r];
    int   bi = bestIdx[r];
    #pragma unroll
    for (int off = 8; off >= 1; off >>= 1) {
      float ov = __shfl_xor(bv, off, 32);
      int   oi = __shfl_xor(bi, off, 32);
      if (ov > bv || (ov == bv && oi < bi)) { bv = ov; bi = oi; }
    }
    if ((lane & 15) == 0) {
      const int m = blockRow + rowFrag + r + ((lane >> 4) ? 8 : 0);
      idx_out[m] = bi;
      ind_f[m]   = (float)bi;
    }
  }
}

// ---------------------------------------------------------------------------
// Kernel 2b (fallback, no workspace planes): inline-convert, single-buffered.
// ---------------------------------------------------------------------------
__global__ void
__launch_bounds__(256)
argmax_fallback_kernel(const float* __restrict__ x,
                       const float* __restrict__ embed,
                       const float* __restrict__ e_sq,
                       int* __restrict__ idx_out,
                       float* __restrict__ ind_f,
                       int K) {
  __shared__ uint16_t sAhi[BM * BKP];
  __shared__ uint16_t sAlo[BM * BKP];
  __shared__ uint16_t sBhi[BN * BKP];
  __shared__ uint16_t sBlo[BN * BKP];

  const int w        = threadIdx.x >> 5;
  const int lane     = threadIdx.x & 31;
  const int rowFrag  = w * 16;
  const int blockRow = blockIdx.x * BM;

  float bestVal[8];
  int   bestIdx[8];
  #pragma unroll
  for (int r = 0; r < 8; ++r) { bestVal[r] = -3.4e38f; bestIdx[r] = 0; }

  for (int nBase = 0; nBase < K; nBase += BN) {
    v8f acc[8];
    #pragma unroll
    for (int t = 0; t < 8; ++t) acc[t] = (v8f){};

    for (int kk = 0; kk < DDIM; kk += BK) {
      __syncthreads();
      load_tile_hilo(x,     blockRow, kk, sAhi, sAlo);
      load_tile_hilo(embed, nBase,    kk, sBhi, sBlo);
      __syncthreads();

      v16bf ah = load_frag(sAhi, rowFrag);
      v16bf al = load_frag(sAlo, rowFrag);
      #pragma unroll
      for (int t = 0; t < 8; ++t) {
        v16bf bh = load_frag(sBhi, t * 16);
        v16bf bl = load_frag(sBlo, t * 16);
        acc[t] = __builtin_amdgcn_wmma_f32_16x16x32_bf16(
            false, al, false, bh, (short)0, acc[t], false, false);
        acc[t] = __builtin_amdgcn_wmma_f32_16x16x32_bf16(
            false, ah, false, bl, (short)0, acc[t], false, false);
        acc[t] = __builtin_amdgcn_wmma_f32_16x16x32_bf16(
            false, ah, false, bh, (short)0, acc[t], false, false);
      }
    }

    #pragma unroll
    for (int t = 0; t < 8; ++t) {
      const int n    = nBase + t * 16 + (lane & 15);
      const float es = e_sq[n];
      #pragma unroll
      for (int r = 0; r < 8; ++r) {
        float s = 2.0f * acc[t][r] - es;
        if (s > bestVal[r]) { bestVal[r] = s; bestIdx[r] = n; }
      }
    }
  }

  #pragma unroll
  for (int r = 0; r < 8; ++r) {
    float bv = bestVal[r];
    int   bi = bestIdx[r];
    #pragma unroll
    for (int off = 8; off >= 1; off >>= 1) {
      float ov = __shfl_xor(bv, off, 32);
      int   oi = __shfl_xor(bi, off, 32);
      if (ov > bv || (ov == bv && oi < bi)) { bv = ov; bi = oi; }
    }
    if ((lane & 15) == 0) {
      const int m = blockRow + rowFrag + r + ((lane >> 4) ? 8 : 0);
      idx_out[m] = bi;
      ind_f[m]   = (float)bi;
    }
  }
}

// ---------------------------------------------------------------------------
// Kernel 3: quantize[row] = embed[idx[row]]   (grid = Nrows, block = 128)
// ---------------------------------------------------------------------------
__global__ void gather_kernel(const float* __restrict__ embed,
                              const int* __restrict__ idx,
                              float* __restrict__ q) {
  const int row = blockIdx.x;
  const int c   = idx[row];
  const float4* src = (const float4*)(embed + (size_t)c * DDIM);
  float4*       dst = (float4*)(q + (size_t)row * DDIM);
  dst[threadIdx.x] = src[threadIdx.x];   // 128 threads x 16B = 512 floats
}

// ---------------------------------------------------------------------------
extern "C" void kernel_launch(void* const* d_in, const int* in_sizes, int n_in,
                              void* d_out, int out_size, void* d_ws, size_t ws_size,
                              hipStream_t stream) {
  const float* x     = (const float*)d_in[0];
  const float* embed = (const float*)d_in[1];
  const int Nrows = in_sizes[0] / DDIM;   // 32768
  const int K     = in_sizes[1] / DDIM;   // 4096

  // Workspace layout
  float* e_sq   = (float*)d_ws;                                    // K floats
  int*   idxbuf = (int*)((char*)d_ws + (size_t)K * sizeof(float)); // Nrows ints
  size_t base   = ((size_t)K * 4 + (size_t)Nrows * 4 + 255) & ~(size_t)255;
  const size_t nx = (size_t)Nrows * DDIM;   // x elems
  const size_t ne = (size_t)K * DDIM;       // embed elems
  uint16_t* xhi = (uint16_t*)((char*)d_ws + base);
  uint16_t* xlo = xhi + nx;
  uint16_t* ehi = xlo + nx;
  uint16_t* elo = ehi + ne;
  const size_t ws_needed = base + 2 * (nx + ne) * sizeof(uint16_t);

  float* out   = (float*)d_out;
  float* ind_f = out;            // argmax indices (as float values)
  float* quant = out + Nrows;    // dequantized vectors

  esq_kernel<<<K, 128, 0, stream>>>(embed, e_sq);

  if (ws_size >= ws_needed) {
    const int n8x = (int)(nx / 8), n8e = (int)(ne / 8);
    convert_hilo_kernel<<<(n8x + 255) / 256, 256, 0, stream>>>(x, xhi, xlo, n8x);
    convert_hilo_kernel<<<(n8e + 255) / 256, 256, 0, stream>>>(embed, ehi, elo, n8e);
    argmax_pre_kernel<<<Nrows / BM, 256, 0, stream>>>(
        xhi, xlo, ehi, elo, e_sq, idxbuf, ind_f, K);
  } else {
    argmax_fallback_kernel<<<Nrows / BM, 256, 0, stream>>>(
        x, embed, e_sq, idxbuf, ind_f, K);
  }

  gather_kernel<<<Nrows, 128, 0, stream>>>(embed, idxbuf, quant);
}
